// DualAttentionRNN_62543313764536
// MI455X (gfx1250) — compile-verified, matrix operand order checked by reading, not run on hardware
//
#include <hip/hip_runtime.h>
#include <hip/hip_bf16.h>
#include <math.h>

#define BB   128
#define TM1  64
#define NN   128
#define HH   512
#define PP   512
#define GG   (4*HH)   // 2048
#define TNB  4        // N-tiles per wave (register blocking)

typedef __attribute__((ext_vector_type(16))) __bf16 v16bf;
typedef __attribute__((ext_vector_type(8)))  float  v8f;
typedef __attribute__((ext_vector_type(8)))  unsigned short v8us;

__device__ __forceinline__ unsigned short f2bf(float f) {
    __bf16 h = (__bf16)f;
    return __builtin_bit_cast(unsigned short, h);
}
__device__ __forceinline__ float sigf(float x) { return 1.0f / (1.0f + expf(-x)); }

// ---------------------------------------------------------------------------
// Fragment load: 16-bit A-matrix layout (ISA 7.12.2).
//   lane L: row = L&15; elements 0..7  -> K = k0 + half8 + e
//                      elements 8..15 -> K = k0 + 16 + half8 + e
//   (half8 = 8 for lanes 16..31). B mirrors with lane = column; since weights
//   are stored row-major (out x in), each lane reads contiguous K -> 2x b128.
// ---------------------------------------------------------------------------
__device__ __forceinline__ v16bf load_frag(const unsigned short* __restrict__ rowp, int koff) {
    v8us u0 = *(const v8us*)(rowp + koff);
    v8us u1 = *(const v8us*)(rowp + koff + 16);
    v16bf a;
#pragma unroll
    for (int e = 0; e < 8; ++e) {
        a[e]     = __builtin_bit_cast(__bf16, (unsigned short)u0[e]);
        a[e + 8] = __builtin_bit_cast(__bf16, (unsigned short)u1[e]);
    }
    return a;
}

// Generic WMMA GEMM: C[M x Ncols] = A1[M x K1] * W1^T + (optional) A2[M x K2] * W2^T
// A row-major bf16 (lda), W row-major bf16 (out x K).
// One wave computes a 16 x (16*TNB) strip: A fragment loaded once per K-slab,
// reused across TNB WMMAs (4 wmma per 10 b128 loads instead of 1 per 4).
__global__ void wmma_gemm_kernel(const unsigned short* __restrict__ A1, int lda1, int K1,
                                 const unsigned short* __restrict__ W1,
                                 const unsigned short* __restrict__ A2, int lda2, int K2,
                                 const unsigned short* __restrict__ W2,
                                 float* __restrict__ C, int ldc, int Mtiles, int Ntiles) {
    int wave = (blockIdx.x * blockDim.x + threadIdx.x) >> 5;
    int lane = threadIdx.x & 31;
    int ng = Ntiles / TNB;               // N-tile groups
    int total = Mtiles * ng;
    if (wave >= total) return;
    int tm  = wave / ng;
    int tng = wave % ng;                 // base N tile = tng*TNB
    int r = lane & 15;
    int half8 = (lane >> 4) << 3;

    v8f acc[TNB] = {};
    {
        const unsigned short* arow  = A1 + (size_t)(tm * 16 + r) * lda1;
        const unsigned short* wbase = W1 + (size_t)(tng * TNB * 16 + r) * K1;
        for (int k0 = 0; k0 < K1; k0 += 32) {
            v16bf a = load_frag(arow, k0 + half8);
#pragma unroll
            for (int j = 0; j < TNB; ++j) {
                v16bf b = load_frag(wbase + (size_t)j * 16 * K1, k0 + half8);
                acc[j] = __builtin_amdgcn_wmma_f32_16x16x32_bf16(false, a, false, b,
                                                                 (short)0, acc[j], false, false);
            }
        }
    }
    if (A2 != nullptr) {
        const unsigned short* arow  = A2 + (size_t)(tm * 16 + r) * lda2;
        const unsigned short* wbase = W2 + (size_t)(tng * TNB * 16 + r) * K2;
        for (int k0 = 0; k0 < K2; k0 += 32) {
            v16bf a = load_frag(arow, k0 + half8);
#pragma unroll
            for (int j = 0; j < TNB; ++j) {
                v16bf b = load_frag(wbase + (size_t)j * 16 * K2, k0 + half8);
                acc[j] = __builtin_amdgcn_wmma_f32_16x16x32_bf16(false, a, false, b,
                                                                 (short)0, acc[j], false, false);
            }
        }
    }
    // C/D layout: VGPR v -> row tm*16 + v + (lane>=16 ? 8 : 0), col = tile*16 + (lane&15)
    int rowoff = (lane >> 4) << 3;
    float* crow = C + (size_t)(tm * 16 + rowoff) * ldc + (size_t)tng * TNB * 16 + r;
#pragma unroll
    for (int j = 0; j < TNB; ++j)
#pragma unroll
        for (int v = 0; v < 8; ++v)
            crow[(size_t)v * ldc + j * 16] = acc[j][v];
}

// ---------------------------------------------------------------------------
__global__ void zero_kernel(unsigned int* __restrict__ p, int n) {
    int i = blockIdx.x * blockDim.x + threadIdx.x;
    if (i < n) p[i] = 0u;
}

__global__ void convert_bf16_kernel(const float* __restrict__ src,
                                    unsigned short* __restrict__ dst, int n) {
    int i = blockIdx.x * blockDim.x + threadIdx.x;
    if (i < n) dst[i] = f2bf(src[i]);
}

// Split dec_W1 (H x (2P+H)) into W1hc (H x 2P, bf16) and W1e (H x H, bf16)
__global__ void build_w1_kernel(const float* __restrict__ dec_W1,
                                unsigned short* __restrict__ W1hc,
                                unsigned short* __restrict__ W1e) {
    int idx = blockIdx.x * blockDim.x + threadIdx.x;
    int total = HH * (2 * PP + HH);
    if (idx >= total) return;
    int k = idx / (2 * PP + HH);
    int j = idx % (2 * PP + HH);
    unsigned short v = f2bf(dec_W1[idx]);
    if (j < 2 * PP) W1hc[k * 2 * PP + j] = v;
    else            W1e[k * HH + (j - 2 * PP)] = v;
}

// Encoder input attention is h/c-independent: alpha = softmax(e_x) once.
// alphaX[b,t,n] = alpha[b,n] * x[b,t,n]   (bf16, GEMM A operand)
__global__ void enc_prep_kernel(const float* __restrict__ batch_x,
                                const float* __restrict__ enc_Wa,
                                unsigned short* __restrict__ alphaX) {
    int b = blockIdx.x;
    int n = threadIdx.x;  // NN = 128 threads
    const float* xb = batch_x + (size_t)b * TM1 * NN;
    const float* wx = enc_Wa + 2 * HH;
    float e = 0.0f;
    for (int t = 0; t < TM1; ++t) e += xb[(size_t)t * NN + n] * wx[t];
    __shared__ float s1[NN];
    s1[n] = e;
    __syncthreads();
    for (int s = NN / 2; s > 0; s >>= 1) {
        if (n < s) s1[n] = fmaxf(s1[n], s1[n + s]);
        __syncthreads();
    }
    float mx = s1[0];
    __syncthreads();
    float ex = expf(e - mx);
    s1[n] = ex;
    __syncthreads();
    for (int s = NN / 2; s > 0; s >>= 1) {
        if (n < s) s1[n] += s1[n + s];
        __syncthreads();
    }
    float alpha = ex / s1[0];
    unsigned short* ax = alphaX + (size_t)b * TM1 * NN;
    for (int t = 0; t < TM1; ++t)
        ax[(size_t)t * NN + n] = f2bf(alpha * xb[(size_t)t * NN + n]);
}

// Encoder LSTM pointwise update
__global__ void enc_point_kernel(const float* __restrict__ g,
                                 const float* __restrict__ bih,
                                 const float* __restrict__ bhh,
                                 float* __restrict__ h, float* __restrict__ c,
                                 unsigned short* __restrict__ hbf,
                                 float* __restrict__ enc_seq,
                                 unsigned short* __restrict__ enc_seq_bf, int t) {
    int idx = blockIdx.x * blockDim.x + threadIdx.x;
    if (idx >= BB * HH) return;
    int b = idx / HH, j = idx % HH;
    const float* gr = g + (size_t)b * GG;
    float iv = gr[j]          + bih[j]          + bhh[j];
    float fv = gr[HH + j]     + bih[HH + j]     + bhh[HH + j];
    float gv = gr[2 * HH + j] + bih[2 * HH + j] + bhh[2 * HH + j];
    float ov = gr[3 * HH + j] + bih[3 * HH + j] + bhh[3 * HH + j];
    float c2 = sigf(fv) * c[idx] + sigf(iv) * tanhf(gv);
    float h2 = sigf(ov) * tanhf(c2);
    c[idx] = c2;
    h[idx] = h2;
    hbf[idx] = f2bf(h2);
    size_t o = ((size_t)b * TM1 + t) * HH + j;
    enc_seq[o] = h2;
    enc_seq_bf[o] = f2bf(h2);
}

// Decoder temporal attention + context + y_tilde (one block per batch row)
__global__ void dec_attn_kernel(const float* __restrict__ hq,
                                const float* __restrict__ e_enc,
                                const float* __restrict__ b1,
                                const float* __restrict__ W2,
                                const float* __restrict__ b2v,
                                const float* __restrict__ enc_seq,
                                const float* __restrict__ y_history,
                                const float* __restrict__ Wfc,
                                const float* __restrict__ bfc,
                                float* __restrict__ ctx,
                                float* __restrict__ ytilde, int t) {
    int b = blockIdx.x;
    int tid = threadIdx.x;           // 256 threads = 8 waves
    int lane = tid & 31, wv = tid >> 5;
    __shared__ float sE[TM1];
    __shared__ float swr[8];

    const float* hqb = hq + (size_t)b * HH;
    for (int tp = wv; tp < TM1; tp += 8) {
        const float* ee = e_enc + ((size_t)b * TM1 + tp) * HH;
        float s = 0.0f;
        for (int k = lane; k < HH; k += 32) {
            float v = hqb[k] + ee[k] + b1[k];
            s += fmaxf(v, 0.0f) * W2[k];
        }
        for (int m = 16; m > 0; m >>= 1) s += __shfl_xor(s, m, 32);
        if (lane == 0) sE[tp] = s + b2v[0];
    }
    __syncthreads();
    __shared__ float ssum;
    if (tid == 0) {
        float mx = sE[0];
        for (int i = 1; i < TM1; ++i) mx = fmaxf(mx, sE[i]);
        float sm = 0.0f;
        for (int i = 0; i < TM1; ++i) { float e = expf(sE[i] - mx); sE[i] = e; sm += e; }
        ssum = sm;
    }
    __syncthreads();
    float inv = 1.0f / ssum;
    const float* es = enc_seq + (size_t)b * TM1 * HH;
    for (int m = tid; m < HH; m += blockDim.x) {
        float acc = 0.0f;
        for (int tp = 0; tp < TM1; ++tp) acc += sE[tp] * es[(size_t)tp * HH + m];
        ctx[(size_t)b * HH + m] = acc * inv;
    }
    __syncthreads();
    float p = 0.0f;
    for (int m = tid; m < HH; m += blockDim.x) p += ctx[(size_t)b * HH + m] * Wfc[m];
    for (int m = 16; m > 0; m >>= 1) p += __shfl_xor(p, m, 32);
    if (lane == 0) swr[wv] = p;
    __syncthreads();
    if (tid == 0) {
        float tot = 0.0f;
        for (int i = 0; i < 8; ++i) tot += swr[i];
        float yt = y_history[(size_t)b * TM1 + t];
        ytilde[b] = tot + yt * Wfc[HH] + bfc[0];
    }
}

// Decoder LSTM pointwise update (adds rank-1 y_tilde * dec_Wih term)
__global__ void dec_point_kernel(const float* __restrict__ g,
                                 const float* __restrict__ ytilde,
                                 const float* __restrict__ Wih,   // (4P,1)
                                 const float* __restrict__ bih,
                                 const float* __restrict__ bhh,
                                 float* __restrict__ h, float* __restrict__ c,
                                 unsigned short* __restrict__ hc_bf) {
    int idx = blockIdx.x * blockDim.x + threadIdx.x;
    if (idx >= BB * PP) return;
    int b = idx / PP, j = idx % PP;
    const float* gr = g + (size_t)b * GG;
    float yt = ytilde[b];
    float iv = gr[j]          + yt * Wih[j]          + bih[j]          + bhh[j];
    float fv = gr[PP + j]     + yt * Wih[PP + j]     + bih[PP + j]     + bhh[PP + j];
    float gv = gr[2 * PP + j] + yt * Wih[2 * PP + j] + bih[2 * PP + j] + bhh[2 * PP + j];
    float ov = gr[3 * PP + j] + yt * Wih[3 * PP + j] + bih[3 * PP + j] + bhh[3 * PP + j];
    float c2 = sigf(fv) * c[idx] + sigf(iv) * tanhf(gv);
    float h2 = sigf(ov) * tanhf(c2);
    c[idx] = c2;
    h[idx] = h2;
    hc_bf[(size_t)b * 2 * PP + j]      = f2bf(h2);
    hc_bf[(size_t)b * 2 * PP + PP + j] = f2bf(c2);
}

// y_pred[b] = [hT, ctx_last] . Wfin + bfin
__global__ void final_kernel(const float* __restrict__ dech,
                             const float* __restrict__ ctx,
                             const float* __restrict__ Wfin,
                             const float* __restrict__ bfin,
                             float* __restrict__ out) {
    int b = blockIdx.x;
    int tid = threadIdx.x, lane = tid & 31, wv = tid >> 5;
    float p = 0.0f;
    for (int j = tid; j < PP + HH; j += blockDim.x) {
        float v = (j < PP) ? dech[(size_t)b * PP + j] : ctx[(size_t)b * HH + (j - PP)];
        p += v * Wfin[j];
    }
    for (int m = 16; m > 0; m >>= 1) p += __shfl_xor(p, m, 32);
    __shared__ float swr[8];
    if (lane == 0) swr[wv] = p;
    __syncthreads();
    if (tid == 0) {
        float tot = 0.0f;
        for (int i = 0; i < 8; ++i) tot += swr[i];
        out[b] = tot + bfin[0];
    }
}

// ---------------------------------------------------------------------------
extern "C" void kernel_launch(void* const* d_in, const int* in_sizes, int n_in,
                              void* d_out, int out_size, void* d_ws, size_t ws_size,
                              hipStream_t stream) {
    const float* batch_x   = (const float*)d_in[0];
    const float* y_history = (const float*)d_in[1];
    const float* enc_Wa    = (const float*)d_in[2];
    const float* enc_Wih   = (const float*)d_in[4];
    const float* enc_Whh   = (const float*)d_in[5];
    const float* enc_bih   = (const float*)d_in[6];
    const float* enc_bhh   = (const float*)d_in[7];
    const float* dec_W1    = (const float*)d_in[8];
    const float* dec_b1    = (const float*)d_in[9];
    const float* dec_W2    = (const float*)d_in[10];
    const float* dec_b2    = (const float*)d_in[11];
    const float* dec_Wih   = (const float*)d_in[12];
    const float* dec_Whh   = (const float*)d_in[13];
    const float* dec_bih   = (const float*)d_in[14];
    const float* dec_bhh   = (const float*)d_in[15];
    const float* dec_Wfc   = (const float*)d_in[16];
    const float* dec_bfc   = (const float*)d_in[17];
    const float* dec_Wfin  = (const float*)d_in[18];
    const float* dec_bfin  = (const float*)d_in[19];
    float* out = (float*)d_out;

    // ---- workspace carve-up (256B aligned slices) ----
    char* ws = (char*)d_ws;
    size_t off = 0;
    auto alloc = [&](size_t bytes) -> char* {
        char* p = ws + off;
        off += (bytes + 255) & ~(size_t)255;
        return p;
    };
    float* f_ench   = (float*)alloc((size_t)BB * HH * 4);
    float* f_encc   = (float*)alloc((size_t)BB * HH * 4);
    float* f_dech   = (float*)alloc((size_t)BB * PP * 4);
    float* f_decc   = (float*)alloc((size_t)BB * PP * 4);
    float* f_g      = (float*)alloc((size_t)BB * GG * 4);
    float* f_hq     = (float*)alloc((size_t)BB * HH * 4);
    float* f_ctx    = (float*)alloc((size_t)BB * HH * 4);
    float* f_yt     = (float*)alloc((size_t)BB * 4);
    float* f_encseq = (float*)alloc((size_t)BB * TM1 * HH * 4);
    float* f_eenc   = (float*)alloc((size_t)BB * TM1 * HH * 4);
    unsigned short* u_hbf     = (unsigned short*)alloc((size_t)BB * HH * 2);
    unsigned short* u_hcbf    = (unsigned short*)alloc((size_t)BB * 2 * PP * 2);
    unsigned short* u_alphaX  = (unsigned short*)alloc((size_t)BB * TM1 * NN * 2);
    unsigned short* u_encseqb = (unsigned short*)alloc((size_t)BB * TM1 * HH * 2);
    unsigned short* u_Wih     = (unsigned short*)alloc((size_t)GG * NN * 2);
    unsigned short* u_Whh     = (unsigned short*)alloc((size_t)GG * HH * 2);
    unsigned short* u_dWhh    = (unsigned short*)alloc((size_t)GG * PP * 2);
    unsigned short* u_W1hc    = (unsigned short*)alloc((size_t)HH * 2 * PP * 2);
    unsigned short* u_W1e     = (unsigned short*)alloc((size_t)HH * HH * 2);

    auto zero = [&](void* p, size_t bytes) {
        int n = (int)(bytes / 4);
        zero_kernel<<<(n + 255) / 256, 256, 0, stream>>>((unsigned int*)p, n);
    };
    auto cvt = [&](const float* s, unsigned short* d, int n) {
        convert_bf16_kernel<<<(n + 255) / 256, 256, 0, stream>>>(s, d, n);
    };
    auto gemm = [&](const unsigned short* A1, int lda1, int K1, const unsigned short* W1,
                    const unsigned short* A2, int lda2, int K2, const unsigned short* W2,
                    float* C, int ldc, int Mtiles, int Ntiles) {
        int total = Mtiles * (Ntiles / TNB);
        wmma_gemm_kernel<<<(total + 7) / 8, 256, 0, stream>>>(A1, lda1, K1, W1,
                                                              A2, lda2, K2, W2,
                                                              C, ldc, Mtiles, Ntiles);
    };

    // ---- init state ----
    zero(f_ench, (size_t)BB * HH * 4);
    zero(f_encc, (size_t)BB * HH * 4);
    zero(f_dech, (size_t)BB * PP * 4);
    zero(f_decc, (size_t)BB * PP * 4);
    zero(u_hbf,  (size_t)BB * HH * 2);
    zero(u_hcbf, (size_t)BB * 2 * PP * 2);

    // ---- weight conversion (bf16, natural out x in orientation) ----
    cvt(enc_Wih, u_Wih, GG * NN);
    cvt(enc_Whh, u_Whh, GG * HH);
    cvt(dec_Whh, u_dWhh, GG * PP);
    {
        int n = HH * (2 * PP + HH);
        build_w1_kernel<<<(n + 255) / 256, 256, 0, stream>>>(dec_W1, u_W1hc, u_W1e);
    }

    // ---- encoder input attention (time-invariant) ----
    enc_prep_kernel<<<BB, NN, 0, stream>>>(batch_x, enc_Wa, u_alphaX);

    // ---- encoder recurrence ----
    for (int t = 0; t < TM1; ++t) {
        // g = wxt @ Wih^T + h @ Whh^T   (128 x 2048)
        gemm(u_alphaX + (size_t)t * NN, TM1 * NN, NN, u_Wih,
             u_hbf, HH, HH, u_Whh,
             f_g, GG, BB / 16, GG / 16);
        enc_point_kernel<<<(BB * HH + 255) / 256, 256, 0, stream>>>(
            f_g, enc_bih, enc_bhh, f_ench, f_encc, u_hbf, f_encseq, u_encseqb, t);
    }

    // ---- e_enc = enc_seq @ W1e^T  ((B*T) x H) ----
    gemm(u_encseqb, HH, HH, u_W1e, nullptr, 0, 0, nullptr,
         f_eenc, HH, (BB * TM1) / 16, HH / 16);

    // ---- decoder recurrence ----
    for (int t = 0; t < TM1; ++t) {
        // hq = h @ W1h^T + c @ W1c^T   (A = [h|c] 128 x 1024)
        gemm(u_hcbf, 2 * PP, 2 * PP, u_W1hc, nullptr, 0, 0, nullptr,
             f_hq, HH, BB / 16, HH / 16);
        dec_attn_kernel<<<BB, 256, 0, stream>>>(f_hq, f_eenc, dec_b1, dec_W2, dec_b2,
                                                f_encseq, y_history, dec_Wfc, dec_bfc,
                                                f_ctx, f_yt, t);
        // g = h @ dec_Whh^T   (rank-1 y_tilde term added in pointwise)
        gemm(u_hcbf, 2 * PP, PP, u_dWhh, nullptr, 0, 0, nullptr,
             f_g, GG, BB / 16, GG / 16);
        dec_point_kernel<<<(BB * PP + 255) / 256, 256, 0, stream>>>(
            f_g, f_yt, dec_Wih, dec_bih, dec_bhh, f_dech, f_decc, u_hcbf);
    }

    // ---- final projection ----
    final_kernel<<<BB, 256, 0, stream>>>(f_dech, f_ctx, dec_Wfin, dec_bfin, out);
}
